// BRTMOE_44762149159149
// MI455X (gfx1250) — matrix operand order repaired and checked by reading, not compiled, not run on hardware
//
#include <hip/hip_runtime.h>
#include <hip/hip_bf16.h>

typedef __attribute__((ext_vector_type(16))) __bf16 v16bf;
typedef __attribute__((ext_vector_type(8)))  __bf16 v8bf;
typedef __attribute__((ext_vector_type(8)))  float  v8f;
typedef __attribute__((ext_vector_type(4)))  unsigned int u32x4;
typedef __attribute__((ext_vector_type(8)))  int          i32x8;
typedef __attribute__((ext_vector_type(4)))  int          i32x4;

static __device__ __forceinline__ __bf16 f2bf(float f) {
    unsigned u = __builtin_bit_cast(unsigned, f);
    u += 0x7FFFu + ((u >> 16) & 1u);           // round-to-nearest-even
    unsigned short h = (unsigned short)(u >> 16);
    return __builtin_bit_cast(__bf16, h);
}

static __device__ __forceinline__ float gelu_tanh(float x) {
    const float k0 = 0.7978845608028654f, k1 = 0.044715f;
    float t = tanhf(k0 * (x + k1 * x * x * x));
    return 0.5f * x * (1.0f + t);
}

// ---------------------------------------------------------------------------
// CDNA5 Tensor Data Mover: DMA a 2D tile (tile_rows x tile_k bf16 elements,
// row stride k_elems) from global memory into LDS at lds_addr (contiguous,
// row-major).  D# bitfields per CDNA5 ISA.
// Issued once per wave; tracked by TENSORcnt.  6-arg builtin form.
// ---------------------------------------------------------------------------
static __device__ __forceinline__ void tdm_load_tile_bf16(
    unsigned lds_addr, const void* gptr, unsigned k_elems,
    unsigned rows, unsigned tile_k, unsigned tile_rows)
{
    const unsigned long long ga = (unsigned long long)(size_t)gptr;
    u32x4 g0;
    g0[0] = 1u;                                     // count=1 (valid user D#)
    g0[1] = lds_addr;                               // lds_addr [63:32]
    g0[2] = (unsigned)(ga & 0xFFFFFFFFu);           // global_addr [95:64]
    g0[3] = (unsigned)((ga >> 32) & 0x1FFFFFFu)     // global_addr [120:96]
          | (2u << 30);                             // type=2 (image) [127:126]
    i32x8 g1;
    g1[0] = (int)(1u << 16);                        // data_size=1 -> 2 bytes
    g1[1] = (int)((k_elems & 0xFFFFu) << 16);       // tensor_dim0 [79:48] lo16
    g1[2] = (int)((k_elems >> 16) & 0xFFFFu)        // tensor_dim0 hi16
          | (int)((rows & 0xFFFFu) << 16);          // tensor_dim1 [111:80] lo16
    g1[3] = (int)((rows >> 16) & 0xFFFFu)           // tensor_dim1 hi16
          | (int)(tile_k << 16);                    // tile_dim0 [127:112]
    g1[4] = (int)tile_rows;                         // tile_dim1; tile_dim2=0
    g1[5] = (int)k_elems;                           // tensor_dim0_stride lo32
    g1[6] = 0;                                      // stride0 hi16, stride1 lo16
    g1[7] = 0;                                      // stride1 hi32
    const i32x4 z4 = {0, 0, 0, 0};                  // groups 2/3: dims>=2 unused
    const i32x8 z8 = {0, 0, 0, 0, 0, 0, 0, 0};
    __builtin_amdgcn_tensor_load_to_lds(g0, g1, z4, z4, z8, 0);
}

// ---------------------------------------------------------------------------
// Convert fp32 [E][R][Cc] -> bf16 [E][Cc][R]  (transpose + downconvert)
// ---------------------------------------------------------------------------
__global__ __launch_bounds__(256) void transpose_cvt_kernel(
    const float* __restrict__ in, __bf16* __restrict__ out, int R, int Cc)
{
    __shared__ float tile[32][33];
    const int e  = blockIdx.z;
    const size_t base = (size_t)e * R * Cc;
    const int c0 = blockIdx.x * 32;
    const int r0 = blockIdx.y * 32;
    const int tx = threadIdx.x & 31;
    const int ty = threadIdx.x >> 5;           // 0..7
    #pragma unroll
    for (int i = 0; i < 32; i += 8)
        tile[ty + i][tx] = in[base + (size_t)(r0 + ty + i) * Cc + (c0 + tx)];
    __syncthreads();
    #pragma unroll
    for (int i = 0; i < 32; i += 8)
        out[base + (size_t)(c0 + ty + i) * R + (r0 + tx)] = f2bf(tile[tx][ty + i]);
}

// ---------------------------------------------------------------------------
// Gating: logits = x @ w_gate, softmax, argmax.  One token per wave32.
// ---------------------------------------------------------------------------
__global__ __launch_bounds__(256) void gate_kernel(
    const float* __restrict__ x, const float* __restrict__ wg,
    int* __restrict__ eidx, float* __restrict__ gate, int D)
{
    const int lane = threadIdx.x & 31;
    const int wave = threadIdx.x >> 5;
    const int s = blockIdx.x * 8 + wave;
    const float* xs = x + (size_t)s * D;

    float acc[8] = {0.f,0.f,0.f,0.f,0.f,0.f,0.f,0.f};
    for (int d = lane; d < D; d += 32) {
        const float xv = xs[d];
        const float* wr = wg + (size_t)d * 8;
        #pragma unroll
        for (int e = 0; e < 8; ++e) acc[e] += xv * wr[e];
    }
    #pragma unroll
    for (int off = 16; off > 0; off >>= 1) {
        #pragma unroll
        for (int e = 0; e < 8; ++e) acc[e] += __shfl_down(acc[e], off);
    }
    if (lane == 0) {
        float m = acc[0]; int best = 0;
        #pragma unroll
        for (int e = 1; e < 8; ++e) if (acc[e] > m) { m = acc[e]; best = e; }
        float sum = 0.f;
        #pragma unroll
        for (int e = 0; e < 8; ++e) sum += __expf(acc[e] - m);
        eidx[s] = best;
        gate[s] = 1.0f / sum;                  // exp(max - max)/sum
    }
}

// ---------------------------------------------------------------------------
// Sequential (token-order) capacity assignment. Single block, chunked scan.
// ---------------------------------------------------------------------------
__global__ __launch_bounds__(256) void assign_slots_kernel(
    const int* __restrict__ eidx, int* __restrict__ token_slot,
    int* __restrict__ slot_token, int S, int E, int C)
{
    __shared__ int ecur[8];
    __shared__ int chunk_e[256];
    const int tid = threadIdx.x;
    for (int i = tid; i < E * C; i += 256) slot_token[i] = -1;
    if (tid < 8) ecur[tid] = 0;
    __syncthreads();
    for (int base = 0; base < S; base += 256) {
        const int s = base + tid;
        const int e = eidx[s];
        chunk_e[tid] = e;
        __syncthreads();
        int rank = 0;
        for (int j = 0; j < tid; ++j) rank += (chunk_e[j] == e);
        const int pos = ecur[e] + rank;
        if (pos < C) { token_slot[s] = e * C + pos; slot_token[e * C + pos] = s; }
        else         { token_slot[s] = -1; }
        __syncthreads();
        if (tid < 8) {
            int cnt = 0;
            for (int j = 0; j < 256; ++j) cnt += (chunk_e[j] == tid);
            ecur[tid] += cnt;
        }
        __syncthreads();
    }
}

// ---------------------------------------------------------------------------
// Dispatch: gather kept tokens into [E][C][D] bf16 (zeros for empty slots)
// ---------------------------------------------------------------------------
__global__ __launch_bounds__(256) void dispatch_kernel(
    const float* __restrict__ x, const int* __restrict__ slot_token,
    __bf16* __restrict__ Xe, int D)
{
    const int slot = blockIdx.x;
    const int s = slot_token[slot];
    __bf16* dst = Xe + (size_t)slot * D;
    if (s < 0) {
        for (int i = threadIdx.x; i < D; i += 256) dst[i] = f2bf(0.f);
        return;
    }
    const float* src = x + (size_t)s * D;
    for (int i = threadIdx.x; i < D; i += 256) dst[i] = f2bf(src[i]);
}

// ---------------------------------------------------------------------------
// Tiled WMMA GEMM:  out[e] = A[e] (MxK, K-contig) * B[e]^T (NxK, K-contig)
// 128x128 block tile, 8 waves, each wave 4x2 tiles of v_wmma_f32_16x16x32_bf16
// Double-buffered LDS fed by the Tensor Data Mover (TENSORcnt-tracked DMA):
// wave 0 issues one tensor_load_to_lds per tile and publishes via barrier.
// MODE 0: out = bf16 GELU(acc) -> H ;  MODE 1: out = fp32 acc -> expert_out
// ---------------------------------------------------------------------------
template <int MODE>
__global__ __launch_bounds__(256) void moe_gemm_bf16_kernel(
    const __bf16* __restrict__ A, const __bf16* __restrict__ B,
    void* __restrict__ Cout, int M, int N, int K)
{
    const int e  = blockIdx.z;
    const int m0 = blockIdx.y * 128;
    const int n0 = blockIdx.x * 128;
    A += (size_t)e * M * K + (size_t)m0 * K;    // tile origin (row m0, k 0)
    B += (size_t)e * N * K + (size_t)n0 * K;    // tile origin (row n0, k 0)

    __shared__ __bf16 As[2][128][32];   // ping-pong buffers, 8 KB each
    __shared__ __bf16 Bs[2][128][32];

    const int tid  = threadIdx.x;
    const int lane = tid & 31;
    const int wave = tid >> 5;
    const int wm   = (wave & 1) * 64;   // wave M offset in tile
    const int wn   = (wave >> 1) * 32;  // wave N offset in tile
    const int half = lane >> 4;         // 0 or 1
    const int l16  = lane & 15;

    // LDS byte offsets of the two tile buffers (low 32 bits of generic addr).
    const unsigned lds_a0 = (unsigned)(size_t)&As[0][0][0];
    const unsigned lds_b0 = (unsigned)(size_t)&Bs[0][0][0];
    const unsigned bufstride = 128u * 32u * 2u;   // 8192 B per buffer

    v8f acc[4][2];
    #pragma unroll
    for (int mt = 0; mt < 4; ++mt)
        #pragma unroll
        for (int nt = 0; nt < 2; ++nt) acc[mt][nt] = (v8f){};

    // Prologue: wave 0 DMAs k-block 0 into buffer 0 (one TDM op per tile).
    if (wave == 0) {
        tdm_load_tile_bf16(lds_a0, A, (unsigned)K, (unsigned)M, 32u, 128u);
        tdm_load_tile_bf16(lds_b0, B, (unsigned)K, (unsigned)N, 32u, 128u);
    }

    int buf = 0;
    for (int kb = 0; kb < K; kb += 32) {
        if (wave == 0) {
            if (kb + 32 < K) {
                const unsigned boff = (unsigned)(buf ^ 1) * bufstride;
                tdm_load_tile_bf16(lds_a0 + boff, A + kb + 32,
                                   (unsigned)K, (unsigned)M, 32u, 128u);
                tdm_load_tile_bf16(lds_b0 + boff, B + kb + 32,
                                   (unsigned)K, (unsigned)N, 32u, 128u);
                // 2 newest TDM ops may stay in flight; stage kb has landed
                __builtin_amdgcn_s_wait_tensorcnt((short)2);
            } else {
                __builtin_amdgcn_s_wait_tensorcnt((short)0);
            }
        }
        __syncthreads();            // publish stage kb to all 8 waves

        const __bf16 (*Asb)[32] = As[buf];
        const __bf16 (*Bsb)[32] = Bs[buf];

        // A fragments: lane half h holds row (l16), K = 16h..16h+15 (contiguous)
        v16bf afrag[4];
        #pragma unroll
        for (int mt = 0; mt < 4; ++mt) {
            const int row = wm + mt * 16 + l16;
            v8bf lo = *(const v8bf*)&Asb[row][16 * half];
            v8bf hi = *(const v8bf*)&Asb[row][16 * half + 8];
            #pragma unroll
            for (int i = 0; i < 8; ++i) { afrag[mt][i] = lo[i]; afrag[mt][i + 8] = hi[i]; }
        }
        // B fragments: lane half h holds col (l16), K = 16h..16h+15 (contiguous)
        v16bf bfrag[2];
        #pragma unroll
        for (int nt = 0; nt < 2; ++nt)
            bfrag[nt] = *(const v16bf*)&Bsb[wn + nt * 16 + l16][16 * half];

        #pragma unroll
        for (int mt = 0; mt < 4; ++mt)
            #pragma unroll
            for (int nt = 0; nt < 2; ++nt)
                acc[mt][nt] = __builtin_amdgcn_wmma_f32_16x16x32_bf16(
                    false, afrag[mt], false, bfrag[nt],
                    (short)0, acc[mt][nt], false, false);

        __syncthreads();            // all waves done reading buf before rewrite
        buf ^= 1;
    }

    // Epilogue (C/D layout: VGPR r -> row r + 8*half, col = l16)
    #pragma unroll
    for (int mt = 0; mt < 4; ++mt) {
        #pragma unroll
        for (int nt = 0; nt < 2; ++nt) {
            const int ncol = n0 + wn + nt * 16 + l16;
            #pragma unroll
            for (int r = 0; r < 8; ++r) {
                const int mrow = m0 + wm + mt * 16 + 8 * half + r;
                if (MODE == 0) {
                    __bf16* H = (__bf16*)Cout + (size_t)e * M * N;
                    H[(size_t)mrow * N + ncol] = f2bf(gelu_tanh(acc[mt][nt][r]));
                } else {
                    float* O = (float*)Cout + (size_t)e * M * N;
                    O[(size_t)mrow * N + ncol] = acc[mt][nt][r];
                }
            }
        }
    }
}

// ---------------------------------------------------------------------------
// Combine: out[s] = gate[s] * expert_out[slot(s)]  (zeros for dropped tokens)
// ---------------------------------------------------------------------------
__global__ __launch_bounds__(256) void combine_kernel(
    const float* __restrict__ Eout, const int* __restrict__ token_slot,
    const float* __restrict__ gate, float* __restrict__ out, int D)
{
    const int s = blockIdx.x;
    const int slot = token_slot[s];
    float* o = out + (size_t)s * D;
    if (slot < 0) {
        for (int i = threadIdx.x; i < D; i += 256) o[i] = 0.f;
        return;
    }
    const float g = gate[s];
    const float* src = Eout + (size_t)slot * D;
    for (int i = threadIdx.x; i < D; i += 256) o[i] = g * src[i];
}

// ---------------------------------------------------------------------------
extern "C" void kernel_launch(void* const* d_in, const int* in_sizes, int n_in,
                              void* d_out, int out_size, void* d_ws, size_t ws_size,
                              hipStream_t stream) {
    const float* x  = (const float*)d_in[0];   // [S, D]
    const float* wg = (const float*)d_in[1];   // [D, E]
    const float* w1 = (const float*)d_in[2];   // [E, D, F]
    const float* w2 = (const float*)d_in[3];   // [E, F, D]
    float* out = (float*)d_out;

    constexpr int S = 8192, D = 2048, E = 8, F = 8192, C = 1024;

    char* ws = (char*)d_ws;
    size_t o = 0;
    __bf16* w1t = (__bf16*)(ws + o); o += (size_t)E * F * D * 2;   // [E][F][D]
    __bf16* w2t = (__bf16*)(ws + o); o += (size_t)E * D * F * 2;   // [E][D][F]
    __bf16* xe  = (__bf16*)(ws + o); o += (size_t)E * C * D * 2;   // [E][C][D]
    __bf16* h   = (__bf16*)(ws + o); o += (size_t)E * C * F * 2;   // [E][C][F]
    float*  eo  = (float* )(ws + o); o += (size_t)E * C * D * 4;   // [E][C][D]
    int* slot_token = (int*)(ws + o); o += (size_t)E * C * 4;
    int* token_slot = (int*)(ws + o); o += (size_t)S * 4;
    float* gate     = (float*)(ws + o); o += (size_t)S * 4;
    int* eidx       = (int*)(ws + o); o += (size_t)S * 4;

    // 1. weight downconvert + transpose (K-contiguous operands for both GEMMs)
    transpose_cvt_kernel<<<dim3(F / 32, D / 32, E), 256, 0, stream>>>(w1, w1t, D, F);
    transpose_cvt_kernel<<<dim3(D / 32, F / 32, E), 256, 0, stream>>>(w2, w2t, F, D);
    // 2. gating
    gate_kernel<<<S / 8, 256, 0, stream>>>(x, wg, eidx, gate, D);
    // 3. ordered capacity assignment
    assign_slots_kernel<<<1, 256, 0, stream>>>(eidx, token_slot, slot_token, S, E, C);
    // 4. dispatch
    dispatch_kernel<<<E * C, 256, 0, stream>>>(x, slot_token, xe, D);
    // 5. expert FFN via WMMA (TDM-fed double-buffered pipeline)
    moe_gemm_bf16_kernel<0><<<dim3(F / 128, C / 128, E), 256, 0, stream>>>(xe, w1t, h, C, F, D);
    moe_gemm_bf16_kernel<1><<<dim3(D / 128, C / 128, E), 256, 0, stream>>>(h, w2t, eo, C, D, F);
    // 6. combine
    combine_kernel<<<S, 256, 0, stream>>>(eo, token_slot, gate, out, D);
}
